// Mamba2Model_82643760710194
// MI455X (gfx1250) — compile-verified, matrix operand order checked by reading, not run on hardware
//
#include <hip/hip_runtime.h>
#include <hip/hip_bf16.h>
#include <math.h>

// ---------------- problem constants (match reference) ----------------
#define BQ     2
#define LQ     2048
#define DQ     1024
#define DINQ   2048
#define NSTATE 16
#define BL     (BQ * LQ)        // 4096 token rows

// ---------------- CDNA5 WMMA types ----------------
typedef __attribute__((ext_vector_type(16))) __bf16 v16bf;
typedef __attribute__((ext_vector_type(8)))  __bf16 v8bf;
typedef __attribute__((ext_vector_type(8)))  float  v8f;

__device__ __forceinline__ unsigned short f32_bf16(float f) {
  unsigned u = __float_as_uint(f);
  u += 0x7fffu + ((u >> 16) & 1u);          // round-to-nearest-even
  return (unsigned short)(u >> 16);
}

// CDNA5 async global->LDS copy (ASYNCcnt-tracked, no VGPR round-trip).
// LDS operand is the low 32 bits of the generic address (ISA: LDS aperture
// truncates generic addr to addr[31:0]).
__device__ __forceinline__ void async_ld_b128(void* lds, const void* gaddr) {
  unsigned ldsa = (unsigned)(uintptr_t)lds;
  asm volatile("global_load_async_to_lds_b128 %0, %1, off"
               :: "v"(ldsa), "v"(gaddr) : "memory");
}
__device__ __forceinline__ void wait_async6() {
  asm volatile("s_wait_asynccnt 0x6" ::: "memory");
}
__device__ __forceinline__ void wait_async0() {
  asm volatile("s_wait_asynccnt 0x0" ::: "memory");
}

// =====================================================================
// add positional embedding: xbuf = x + pos_emb  (never mutate d_in)
// =====================================================================
__global__ void add_pos_kernel(const float* __restrict__ x,
                               const float* __restrict__ pe,
                               float* __restrict__ xbuf) {
  int idx = blockIdx.x * 256 + threadIdx.x;       // over B*L*D
  if (idx >= BL * DQ) return;
  int ld = idx % (LQ * DQ);
  xbuf[idx] = x[idx] + pe[ld];
}

// =====================================================================
// weight transpose + fp32->bf16:  W[K][N] fp32  ->  WT[N][K] bf16
// =====================================================================
__global__ __launch_bounds__(256) void wconv_kernel(
    const float* __restrict__ W, unsigned short* __restrict__ WT,
    int K, int N) {
  __shared__ float tile[32][33];
  int nb = blockIdx.x * 32, kb = blockIdx.y * 32;
  int tx = threadIdx.x & 31, ty0 = threadIdx.x >> 5;
  for (int r = ty0; r < 32; r += 8)
    tile[r][tx] = W[(size_t)(kb + r) * N + nb + tx];
  __syncthreads();
  for (int r = ty0; r < 32; r += 8)
    WT[(size_t)(nb + r) * K + kb + tx] = f32_bf16(tile[tx][r]);
}

// =====================================================================
// LayerNorm over last dim; optional bf16 output (GEMM A operand)
// =====================================================================
template <bool BF16OUT>
__global__ __launch_bounds__(256) void ln_kernel(
    const float* __restrict__ x, const float* __restrict__ w,
    const float* __restrict__ b, void* __restrict__ out, int Dn) {
  int row = blockIdx.x;
  const float* xr = x + (size_t)row * Dn;
  __shared__ float red[256];
  float s = 0.f, s2 = 0.f;
  for (int i = threadIdx.x; i < Dn; i += 256) {
    float v = xr[i]; s += v; s2 += v * v;
  }
  red[threadIdx.x] = s; __syncthreads();
  for (int o = 128; o > 0; o >>= 1) {
    if (threadIdx.x < o) red[threadIdx.x] += red[threadIdx.x + o];
    __syncthreads();
  }
  float mean = red[0] / (float)Dn;
  __syncthreads();
  red[threadIdx.x] = s2; __syncthreads();
  for (int o = 128; o > 0; o >>= 1) {
    if (threadIdx.x < o) red[threadIdx.x] += red[threadIdx.x + o];
    __syncthreads();
  }
  float var  = red[0] / (float)Dn - mean * mean;
  float rstd = rsqrtf(var + 1e-5f);
  for (int i = threadIdx.x; i < Dn; i += 256) {
    float v = (xr[i] - mean) * rstd * w[i] + b[i];
    if (BF16OUT) ((unsigned short*)out)[(size_t)row * Dn + i] = f32_bf16(v);
    else         ((float*)out)[(size_t)row * Dn + i] = v;
  }
}

// =====================================================================
// WMMA bf16 GEMM:  out[M][N] = A[M][K](bf16) * BT[N][K](bf16)^T + bias
// Block tile 128x64, 8 waves (wave32) arranged 4x2, each wave 32x32 via
// 2x2 v_wmma_f32_16x16x32_bf16 tiles. K staged in 64-wide chunks using
// DOUBLE-BUFFERED async global->LDS copies (ASYNCcnt pipeline):
//   issue stage i+1 into buf^1, s_wait_asynccnt 6 (stage i oldest),
//   barrier, 8 WMMAs on buf, barrier.
// Epilogues:
//   MODE 0: outF = v+bias (fp32) AND outBf = bf16(v+bias)
//   MODE 1: outF = softplus(v+bias)
//   MODE 2: outF = v+bias+outF  (residual add in place)
// =====================================================================
#define BMT 128
#define BNT 64
#define KC  64
#define LDA 72   // padded row stride (elements): 144B = 9*16B

__device__ __forceinline__ v16bf load_frag(const unsigned short* rowp, int half) {
  // A-fragment lane layout (ISA 7.12.2, 16-bit A 16x32):
  //  lanes 0-15:  elems[0..7]=K0..7,  elems[8..15]=K16..23
  //  lanes 16-31: elems[0..7]=K8..15, elems[8..15]=K24..31
  union { v16bf v; v8bf h[2]; } u;
  u.h[0] = *(const v8bf*)(rowp + half * 8);
  u.h[1] = *(const v8bf*)(rowp + 16 + half * 8);
  return u.v;
}

template <int MODE>
__global__ __launch_bounds__(256) void gemm_bf16_wmma(
    const unsigned short* __restrict__ A,   // [M][K] bf16
    const unsigned short* __restrict__ BT,  // [N][K] bf16 (B transposed)
    const float* __restrict__ bias,         // [N]
    float* __restrict__ outF,
    unsigned short* __restrict__ outBf,
    int M, int Nn, int K) {
  __shared__ __align__(16) unsigned short As[2][BMT][LDA];
  __shared__ __align__(16) unsigned short Bs[2][BNT][LDA];
  int tid  = threadIdx.x;
  int wave = tid >> 5, lane = tid & 31;
  int wm = wave & 3, wn = wave >> 2;          // 4 waves along M, 2 along N
  int bm = blockIdx.y * BMT, bn = blockIdx.x * BNT;
  int half = lane >> 4, lrow = lane & 15;

  // per-thread source/dest for one stage: 4 A-chunks + 2 B-chunks of 16B
  int ra[4], ca[4], rb[2], cb[2];
#pragma unroll
  for (int t = 0; t < 4; ++t) {
    int idx = tid + t * 256;                  // 0..1023 -> 128 rows x 8 chunks
    ra[t] = idx >> 3; ca[t] = (idx & 7) * 8;
  }
#pragma unroll
  for (int t = 0; t < 2; ++t) {
    int idx = tid + t * 256;                  // 0..511 -> 64 rows x 8 chunks
    rb[t] = idx >> 3; cb[t] = (idx & 7) * 8;
  }

  auto issue_stage = [&](int buf, int k0) {
#pragma unroll
    for (int t = 0; t < 4; ++t)
      async_ld_b128(&As[buf][ra[t]][ca[t]],
                    A + (size_t)(bm + ra[t]) * K + k0 + ca[t]);
#pragma unroll
    for (int t = 0; t < 2; ++t)
      async_ld_b128(&Bs[buf][rb[t]][cb[t]],
                    BT + (size_t)(bn + rb[t]) * K + k0 + cb[t]);
  };

  v8f zero = {};
  v8f acc[2][2];
#pragma unroll
  for (int i = 0; i < 2; ++i)
#pragma unroll
    for (int j = 0; j < 2; ++j) acc[i][j] = zero;

  const int nk = K / KC;
  issue_stage(0, 0);
  for (int i = 0; i < nk; ++i) {
    int cur = i & 1;
    if (i + 1 < nk) {
      issue_stage(cur ^ 1, (i + 1) * KC);
      wait_async6();                 // stage i's 6 ops are the oldest
    } else {
      wait_async0();
    }
    __syncthreads();                 // all waves' stage-i data visible

#pragma unroll
    for (int ks = 0; ks < KC; ks += 32) {
      v16bf aF[2], bF[2];
      aF[0] = load_frag(&As[cur][wm * 32 + lrow][ks],      half);
      aF[1] = load_frag(&As[cur][wm * 32 + 16 + lrow][ks], half);
      bF[0] = load_frag(&Bs[cur][wn * 32 + lrow][ks],      half);
      bF[1] = load_frag(&Bs[cur][wn * 32 + 16 + lrow][ks], half);
#pragma unroll
      for (int ii = 0; ii < 2; ++ii)
#pragma unroll
        for (int jj = 0; jj < 2; ++jj)
          acc[ii][jj] = __builtin_amdgcn_wmma_f32_16x16x32_bf16(
              false, aF[ii], false, bF[jj], (short)0, acc[ii][jj],
              false, false);
    }
    __syncthreads();                 // buf reusable two stages later
  }

  // ---- epilogue: C/D layout — n = lane&15, m = vgpr + (lane>>4)*8 ----
#pragma unroll
  for (int i = 0; i < 2; ++i) {
#pragma unroll
    for (int j = 0; j < 2; ++j) {
      int gn = bn + wn * 32 + j * 16 + lrow;
      float bv = bias[gn];
#pragma unroll
      for (int r = 0; r < 8; ++r) {
        int gm = bm + wm * 32 + i * 16 + r + half * 8;
        size_t off = (size_t)gm * Nn + gn;
        float v = acc[i][j][r] + bv;
        if (MODE == 0) {
          outF[off]  = v;
          outBf[off] = f32_bf16(v);
        } else if (MODE == 1) {
          outF[off] = (v > 20.f) ? v : log1pf(__expf(v));   // softplus
        } else {
          outF[off] = v + outF[off];                        // + residual
        }
      }
    }
  }
}

// =====================================================================
// Bm|Cm projection:  BC[m][0..15]=h@WB, [16..31]=h@WC   (~0.8% of FLOPs)
// =====================================================================
__global__ __launch_bounds__(256) void bc_kernel(
    const float* __restrict__ h, const float* __restrict__ WB,
    const float* __restrict__ WC, float* __restrict__ BC) {
  int m = blockIdx.x;
  __shared__ float hrow[DINQ];
  __shared__ float part[256];
  for (int i = threadIdx.x; i < DINQ; i += 256)
    hrow[i] = h[(size_t)m * DINQ + i];
  __syncthreads();
  int c = threadIdx.x & 31, g = threadIdx.x >> 5;
  const float* W = (c < 16) ? WB : WC;
  int cc = c & 15;
  float s = 0.f;
  int k0 = g * (DINQ / 8);
  for (int k = k0; k < k0 + DINQ / 8; ++k)
    s += hrow[k] * W[(size_t)k * NSTATE + cc];
  part[threadIdx.x] = s; __syncthreads();
  if (threadIdx.x < 32) {
    float t = 0.f;
#pragma unroll
    for (int g2 = 0; g2 < 8; ++g2) t += part[threadIdx.x + 32 * g2];
    BC[(size_t)m * 32 + threadIdx.x] = t;
  }
}

// =====================================================================
// causal depthwise conv (K=4) + SiLU
// =====================================================================
__global__ void conv_silu_kernel(const float* __restrict__ h,
                                 const float* __restrict__ cw,
                                 const float* __restrict__ cb,
                                 float* __restrict__ yconv) {
  int idx = blockIdx.x * 256 + threadIdx.x;      // over B*L*DIN
  if (idx >= BL * DINQ) return;
  int d = idx % DINQ;
  int l = (idx / DINQ) % LQ;
  int b = idx / (DINQ * LQ);
  float acc = cb[d];
#pragma unroll
  for (int j = 0; j < 4; ++j) {
    int lt = l - 3 + j;
    if (lt >= 0) acc += h[((size_t)b * LQ + lt) * DINQ + d] * cw[j * DINQ + d];
  }
  yconv[idx] = acc * (1.f / (1.f + __expf(-acc)));   // silu
}

// =====================================================================
// selective scan (diagonal SSM), fused D-skip + sigmoid gating.
// 4 lanes per channel (4 states each), y reduced with wave32 shfl_xor.
// B/C vectors (shared by all channels) staged via LDS in 32-step chunks.
// =====================================================================
#define TCH 32
__global__ __launch_bounds__(256) void scan_kernel(
    const float* __restrict__ h,  const float* __restrict__ dt,
    const float* __restrict__ BC, const float* __restrict__ yconv,
    const float* __restrict__ A_log, const float* __restrict__ Dsk,
    unsigned short* __restrict__ gout) {
  const int q    = threadIdx.x & 3;          // state sub-group 0..3
  const int dloc = threadIdx.x >> 2;         // 0..63 channels per block
  const int blocks_per_b = DINQ / 64;        // 32
  const int b = blockIdx.x / blocks_per_b;
  const int d = (blockIdx.x % blocks_per_b) * 64 + dloc;

  float a[4], hs[4];
#pragma unroll
  for (int n2 = 0; n2 < 4; ++n2) {
    a[n2]  = -__expf(A_log[(size_t)d * NSTATE + q * 4 + n2]);
    hs[n2] = 0.f;
  }
  const float dsk = Dsk[d];
  __shared__ float bcS[TCH][32];

  for (int t0 = 0; t0 < LQ; t0 += TCH) {
    __syncthreads();
    for (int i = threadIdx.x; i < TCH * 32; i += 256) {
      int tt = i >> 5, j = i & 31;
      bcS[tt][j] = BC[(size_t)(b * LQ + t0 + tt) * 32 + j];
    }
    __syncthreads();
    for (int tt = 0; tt < TCH; ++tt) {
      size_t off = ((size_t)b * LQ + t0 + tt) * DINQ + d;
      float dtv = dt[off];
      float xv  = h[off];
      float dx  = dtv * xv;
      float y = 0.f;
#pragma unroll
      for (int n2 = 0; n2 < 4; ++n2) {
        int n = q * 4 + n2;
        float dA = __expf(dtv * a[n2]);
        hs[n2] = hs[n2] * dA + dx * bcS[tt][n];
        y += hs[n2] * bcS[tt][16 + n];
      }
      y += __shfl_xor(y, 1);
      y += __shfl_xor(y, 2);
      if (q == 0) {
        y += xv * dsk;
        float yc = yconv[off];
        gout[off] = f32_bf16(y * (1.f / (1.f + __expf(-yc))));
      }
    }
  }
}

// =====================================================================
// host-side orchestration
// =====================================================================
extern "C" void kernel_launch(void* const* d_in, const int* in_sizes, int n_in,
                              void* d_out, int out_size, void* d_ws, size_t ws_size,
                              hipStream_t stream) {
  const float* x_in  = (const float*)d_in[0];
  const float* pos   = (const float*)d_in[1];
  const float* normw = (const float*)d_in[2];
  const float* normb = (const float*)d_in[3];
  const float* Win   = (const float*)d_in[4];
  const float* b_in  = (const float*)d_in[5];
  const float* A_log = (const float*)d_in[6];
  const float* Wdt   = (const float*)d_in[7];
  const float* b_dt  = (const float*)d_in[8];
  const float* WB    = (const float*)d_in[9];
  const float* WC    = (const float*)d_in[10];
  const float* Dsk   = (const float*)d_in[11];
  const float* convw = (const float*)d_in[12];
  const float* convb = (const float*)d_in[13];
  const float* Wout  = (const float*)d_in[14];
  const float* b_out = (const float*)d_in[15];
  const float* fnw   = (const float*)d_in[16];
  const float* fnb   = (const float*)d_in[17];
  (void)in_sizes; (void)n_in; (void)out_size; (void)ws_size;

  char* p = (char*)d_ws;
  auto alloc = [&](size_t bytes) {
    char* r = p;
    p += (bytes + 255) & ~(size_t)255;
    return r;
  };
  float*          xbuf  = (float*)         alloc((size_t)BL * DQ   * 4);  // residual stream
  unsigned short* hln   = (unsigned short*)alloc((size_t)BL * DQ   * 2);  // LN(x) bf16
  float*          hbuf  = (float*)         alloc((size_t)BL * DINQ * 4);  // h fp32
  unsigned short* hbf   = (unsigned short*)alloc((size_t)BL * DINQ * 2);  // h bf16
  float*          dtb   = (float*)         alloc((size_t)BL * DINQ * 4);  // softplus(dt)
  float*          BCb   = (float*)         alloc((size_t)BL * 32   * 4);  // [B|C]
  float*          yconv = (float*)         alloc((size_t)BL * DINQ * 4);  // silu(conv)
  unsigned short* gbf   = (unsigned short*)alloc((size_t)BL * DINQ * 2);  // gated, bf16
  unsigned short* WinT  = (unsigned short*)alloc((size_t)DINQ * DQ   * 2);
  unsigned short* WdtT  = (unsigned short*)alloc((size_t)DINQ * DINQ * 2);
  unsigned short* WoutT = (unsigned short*)alloc((size_t)DQ   * DINQ * 2);

  add_pos_kernel<<<(BL * DQ + 255) / 256, 256, 0, stream>>>(x_in, pos, xbuf);

  for (int i = 0; i < 2; ++i) {
    wconv_kernel<<<dim3(DINQ / 32, DQ / 32),   256, 0, stream>>>(
        Win  + (size_t)i * DQ * DINQ,   WinT,  DQ,   DINQ);
    wconv_kernel<<<dim3(DINQ / 32, DINQ / 32), 256, 0, stream>>>(
        Wdt  + (size_t)i * DINQ * DINQ, WdtT,  DINQ, DINQ);
    wconv_kernel<<<dim3(DQ / 32, DINQ / 32),   256, 0, stream>>>(
        Wout + (size_t)i * DINQ * DQ,   WoutT, DINQ, DQ);

    ln_kernel<true><<<BL, 256, 0, stream>>>(
        xbuf, normw + (size_t)i * DQ, normb + (size_t)i * DQ, hln, DQ);

    // h = LN(x) @ Win + b_in    (fp32 + bf16 copies)
    gemm_bf16_wmma<0><<<dim3(DINQ / BNT, BL / BMT), 256, 0, stream>>>(
        hln, WinT, b_in + (size_t)i * DINQ, hbuf, hbf, BL, DINQ, DQ);
    // dt = softplus(h @ Wdt + b_dt)
    gemm_bf16_wmma<1><<<dim3(DINQ / BNT, BL / BMT), 256, 0, stream>>>(
        hbf, WdtT, b_dt + (size_t)i * DINQ, dtb, nullptr, BL, DINQ, DINQ);

    bc_kernel<<<BL, 256, 0, stream>>>(
        hbuf, WB + (size_t)i * DINQ * NSTATE, WC + (size_t)i * DINQ * NSTATE, BCb);
    conv_silu_kernel<<<(BL * DINQ + 255) / 256, 256, 0, stream>>>(
        hbuf, convw + (size_t)i * 4 * DINQ, convb + (size_t)i * DINQ, yconv);
    scan_kernel<<<BQ * (DINQ / 64), 256, 0, stream>>>(
        hbuf, dtb, BCb, yconv, A_log + (size_t)i * DINQ * NSTATE,
        Dsk + (size_t)i * DINQ, gbf);

    // x = g @ Wout + b_out + residual (in place)
    gemm_bf16_wmma<2><<<dim3(DQ / BNT, BL / BMT), 256, 0, stream>>>(
        gbf, WoutT, b_out + (size_t)i * DQ, xbuf, nullptr, BL, DQ, DINQ);
  }

  ln_kernel<false><<<BL, 256, 0, stream>>>(xbuf, fnw, fnb, d_out, DQ);
}
